// CorrTorch_lkm_53171695124498
// MI455X (gfx1250) — compile-verified, matrix operand order checked by reading, not run on hardware
//
#include <hip/hip_runtime.h>

#define B_   8
#define C_   256
#define H_   96
#define W_   128
#define NDY  9

#define STR   40           // LDS row stride in ushorts (80 B, 16B-aligned rows)
#define BROWS 144          // in2 LDS cols: global cols -4 .. 139

typedef __attribute__((ext_vector_type(16))) __bf16 v16bf;
typedef __attribute__((ext_vector_type(8)))  float  v8f;

// Pack two floats into one dword of two bf16 (k in [15:0], k+1 in [31:16]).
// Pin the operand pairing to a single v_cvt_pk_bf16_f32 (the SLP vectorizer
// otherwise packs in load order and re-interleaves with ~8 extra VALU ops).
__device__ __forceinline__ unsigned pack_bf16(float lo, float hi) {
  unsigned r;
  asm("v_cvt_pk_bf16_f32 %0, %1, %2" : "=v"(r) : "v"(lo), "v"(hi));
  return r;
}

union Frag { v16bf v; uint4 q[2]; };

// Stage one K-chunk (32 channels) of in1 row and padded in2 row into LDS,
// transposed to [w][k] bf16 so WMMA fragments are contiguous per lane.
__device__ __forceinline__ void fill_chunk(unsigned short* __restrict__ la,
                                           unsigned short* __restrict__ lb,
                                           const float* __restrict__ p1,
                                           const float* __restrict__ p2,
                                           int c0, int tid, size_t cstride) {
  // in1: 16 k-pairs x 32 w-quads = 512 tasks, 2 per thread (coalesced b128).
#pragma unroll
  for (int it = 0; it < 2; ++it) {
    const int t  = tid + (it << 8);
    const int k  = (t >> 5) << 1;        // 0,2,..,30
    const int w4 = (t & 31) << 2;        // 0,4,..,124
    const float* s0 = p1 + (size_t)(c0 + k) * cstride + w4;
    const float4 u0 = *(const float4*)s0;
    const float4 u1 = *(const float4*)(s0 + cstride);
    *(unsigned*)(la + (w4 + 0) * STR + k) = pack_bf16(u0.x, u1.x);
    *(unsigned*)(la + (w4 + 1) * STR + k) = pack_bf16(u0.y, u1.y);
    *(unsigned*)(la + (w4 + 2) * STR + k) = pack_bf16(u0.z, u1.z);
    *(unsigned*)(la + (w4 + 3) * STR + k) = pack_bf16(u0.w, u1.w);
  }

  // in2: 16 k-pairs x 37 col-quads (cols -4..143); quad q covers cols
  // 4q-4..4q-1, fully in-image iff 1 <= q <= 32, else zero-filled.
  for (int t = tid; t < 16 * 37; t += 256) {
    const int kp = t / 37;
    const int q  = t - kp * 37;          // 0..36
    const int k  = kp << 1;
    const bool valid = (q >= 1) && (q <= 32);
    const int  qc = valid ? q : 1;       // clamp: always-legal address
    const float* s0 = p2 + (size_t)(c0 + k) * cstride + (qc * 4 - 4);
    float4 u0 = *(const float4*)s0;
    float4 u1 = *(const float4*)(s0 + cstride);
    if (!valid) {
      u0.x = u0.y = u0.z = u0.w = 0.0f;
      u1 = u0;
    }
    const int wb = q << 2;               // LDS col index = global col + 4
    *(unsigned*)(lb + (wb + 0) * STR + k) = pack_bf16(u0.x, u1.x);
    *(unsigned*)(lb + (wb + 1) * STR + k) = pack_bf16(u0.y, u1.y);
    *(unsigned*)(lb + (wb + 2) * STR + k) = pack_bf16(u0.z, u1.z);
    *(unsigned*)(lb + (wb + 3) * STR + k) = pack_bf16(u0.w, u1.w);
  }
}

// One K-chunk of WMMA: 2 x ds_load_b128 per fragment, 2 accumulating WMMAs.
__device__ __forceinline__ void compute_chunk(const unsigned short* la,
                                              const unsigned short* lb,
                                              int w0, int nl, int khalf,
                                              v8f& acc1, v8f& acc2) {
  Frag fa, fb1, fb2;
  const unsigned short* pa = la + (w0 + nl) * STR + khalf;
  fa.q[0] = *(const uint4*)pa;            // k = khalf .. khalf+7
  fa.q[1] = *(const uint4*)(pa + 16);     // k = 16+khalf .. +7

  const unsigned short* pb1 = lb + (w0 + nl) * STR + khalf;       // col w0-4+nl
  fb1.q[0] = *(const uint4*)pb1;
  fb1.q[1] = *(const uint4*)(pb1 + 16);

  const unsigned short* pb2 = lb + (w0 + 16 + nl) * STR + khalf;  // col w0+12+nl
  fb2.q[0] = *(const uint4*)pb2;
  fb2.q[1] = *(const uint4*)(pb2 + 16);

  acc1 = __builtin_amdgcn_wmma_f32_16x16x32_bf16(false, fa.v, false, fb1.v,
                                                 (short)0, acc1, false, false);
  acc2 = __builtin_amdgcn_wmma_f32_16x16x32_bf16(false, fa.v, false, fb2.v,
                                                 (short)0, acc2, false, false);
}

__global__ __launch_bounds__(256)
void corr_wmma_kernel(const float* __restrict__ in1,
                      const float* __restrict__ in2,
                      float* __restrict__ out) {
  __shared__ __align__(16) unsigned short lds_a[2][W_ * STR];     // [w][k]
  __shared__ __align__(16) unsigned short lds_b[2][BROWS * STR];  // [col+4][k]

  const int h    = blockIdx.x;        // 0..95
  const int dy   = blockIdx.y;        // 0..8
  const int b    = blockIdx.z;        // 0..7
  const int tid  = threadIdx.x;
  const int wave = tid >> 5;          // 0..7 -> W-tile
  const int lane = tid & 31;

  const int w0    = wave << 4;
  const int nl    = lane & 15;        // M (A frag) / N (B frag)
  const int khalf = (lane >> 4) << 3; // 0 or 8 (K sub-block per 16-bit layout)

  const int  h2     = h + dy - 4;
  const bool hvalid = (h2 >= 0) && (h2 < H_);

  v8f acc1 = {};   // N window: in2 cols w0-4 .. w0+11
  v8f acc2 = {};   // N window: in2 cols w0+12 .. w0+27

  const size_t cstride = (size_t)H_ * W_;
  const float* p1 = in1 + ((size_t)b * C_ * H_ + (size_t)h) * W_;
  const float* p2 = in2 + ((size_t)b * C_ * H_ + (size_t)(hvalid ? h2 : 0)) * W_;

  if (hvalid) {
    // Software pipeline over 8 K-chunks, double-buffered LDS:
    //   fill(i+1) overlaps compute(i); one barrier per chunk.
    fill_chunk(lds_a[0], lds_b[0], p1, p2, 0, tid, cstride);
    __syncthreads();
#pragma unroll
    for (int i = 0; i < 8; ++i) {
      if (i < 7) {
        fill_chunk(lds_a[(i + 1) & 1], lds_b[(i + 1) & 1], p1, p2,
                   32 * (i + 1), tid, cstride);
      }
      compute_chunk(lds_a[i & 1], lds_b[i & 1], w0, nl, khalf, acc1, acc2);
      __syncthreads();
    }
  }

  // ---- Band extraction + store (writes zeros when hvalid == false).
  const float scale = 1.0f / (float)C_;
  const int   moff  = (lane >> 4) << 3;      // C/D layout: lanes 16..31 hold M+8
#pragma unroll
  for (int r = 0; r < 8; ++r) {
    const int m = r + moff;                  // pixel within tile, 0..15
    const int w = w0 + m;                    // always < W
    const int dx1 = nl - m;                  // tile 1: col n=nl -> in2 col w0-4+nl
    if (dx1 >= 0 && dx1 <= 8) {
      out[(((size_t)b * 81 + (size_t)(dy * 9 + dx1)) * H_ + h) * W_ + w] =
          acc1[r] * scale;
    }
    const int dx2 = nl + 16 - m;             // tile 2: col n=nl -> in2 col w0+12+nl
    if (dx2 <= 8) {
      out[(((size_t)b * 81 + (size_t)(dy * 9 + dx2)) * H_ + h) * W_ + w] =
          acc2[r] * scale;
    }
  }
}

extern "C" void kernel_launch(void* const* d_in, const int* in_sizes, int n_in,
                              void* d_out, int out_size, void* d_ws, size_t ws_size,
                              hipStream_t stream) {
  const float* in1 = (const float*)d_in[0];
  const float* in2 = (const float*)d_in[1];
  float* out = (float*)d_out;

  dim3 grid(H_, NDY, B_);   // 96 x 9 x 8 = 6912 blocks
  dim3 block(256);          // 8 waves; wave i owns W-tile [16i, 16i+16)
  hipLaunchKernelGGL(corr_wmma_kernel, grid, block, 0, stream, in1, in2, out);
}
